// ResidualClassifier_27376121544992
// MI455X (gfx1250) — matrix operand order repaired and checked by reading, not compiled, not run on hardware
//
#include <hip/hip_runtime.h>

#define EPSF 1e-12f

typedef __attribute__((ext_vector_type(16))) _Float16 v16h;
typedef __attribute__((ext_vector_type(8)))  float    v8f;

union Frag16   { v16h v; _Float16 h[16]; };
union Frag8f   { v8f  v; float    f[8];  };
union HalfPair { unsigned int u; _Float16 h[2]; };

// ---------------------------------------------------------------------------
// Layer 0: Cin=1 outer product. z0[j][n][d] = mean4(x[n, adj0[j,:]]) * w0[j,d] + b0[j,d]
// One block of 128 threads per node (65536 nodes). Output stored as f16
// (packed half-pairs -> b32 stores) to halve the dominant z0 traffic.
// ---------------------------------------------------------------------------
__global__ __launch_bounds__(128) void fgl0_kernel(
    const float* __restrict__ x, const int* __restrict__ adj,
    const float* __restrict__ v, const float* __restrict__ g,
    const float* __restrict__ b, _Float16* __restrict__ zout)
{
  constexpr int NIN = 212445;
  const int node = blockIdx.x;
  const int tid  = threadIdx.x;
  __shared__ float s_h[32];
  __shared__ float s_w[32];
  __shared__ float s_b[32];
  __shared__ int   s_adj[4];
  if (tid < 4) s_adj[tid] = adj[node * 4 + tid];
  __syncthreads();
  if (tid < 32) {
    const float* xr = x + (size_t)tid * NIN;
    s_h[tid] = 0.25f * (xr[s_adj[0]] + xr[s_adj[1]] + xr[s_adj[2]] + xr[s_adj[3]]);
  } else if (tid < 64) {
    const int d = tid - 32;
    const float vv = v[(size_t)node * 32 + d];
    s_w[d] = g[(size_t)node * 32 + d] * vv * rsqrtf(vv * vv + EPSF);
    s_b[d] = b[(size_t)node * 32 + d];
  }
  __syncthreads();
  unsigned int* zo = (unsigned int*)(zout + (size_t)node * 1024);
  #pragma unroll
  for (int i = 0; i < 4; ++i) {
    const int p = tid + i * 128;            // pair index (512 pairs/node)
    const int e = p * 2;                    // e = n*32 + d
    const int n = e >> 5, d = e & 31;
    HalfPair hp;
    hp.h[0] = (_Float16)(s_h[n] * s_w[d]     + s_b[d]);
    hp.h[1] = (_Float16)(s_h[n] * s_w[d + 1] + s_b[d + 1]);
    zo[p] = hp.u;
  }
}

// ---------------------------------------------------------------------------
// Generic FGL layer on WMMA: per output node j,
//   H[32 x CIN] = mean over DEG gathered f16 [32 x CIN] blocks of zin (or self)
//   W[CIN x COUT] = weight-norm(v[j]) (column-wise), converted to f16
//   zout[j] = H @ W + b[j]  (+ H if ADDIN: fused residual, read back from LDS)
// Block = 2*COUT threads = COUT/16 waves; wave w owns N-tile w, both M-tiles,
// iterating CIN/32 K-chunks of v_wmma_f32_16x16x32_f16. TOUT: f16 or f32.
// ---------------------------------------------------------------------------
template<int CIN, int COUT, int DEG, bool SELF, bool ADDIN, typename TOUT>
__global__ __launch_bounds__(COUT * 2) void fgl_wmma_kernel(
    const _Float16* __restrict__ zin, const int* __restrict__ adj,
    const float* __restrict__ v, const float* __restrict__ g,
    const float* __restrict__ b, TOUT* __restrict__ zout)
{
  static_assert(!ADDIN || (CIN == COUT), "residual requires CIN==COUT");
  constexpr int NTHR = COUT * 2;       // NTHR % COUT == 0 (thread<->column fixed)
  constexpr int PAD  = 8;              // keeps rows 16B-aligned, spreads banks
  constexpr int KCH  = CIN / 32;       // K-chunks of 32 per WMMA
  constexpr int HN   = 32 * CIN;
  constexpr int HNP  = HN / 2;         // half-pairs per block of zin

  const int node = blockIdx.x;
  const int tid  = threadIdx.x;

  __shared__ _Float16 Wt[COUT][CIN + PAD];   // transposed weights: Wt[d][c]
  __shared__ _Float16 Hh[32][CIN + PAD];     // activations (f16)
  __shared__ float    s_sum[COUT];
  __shared__ float    s_scale[COUT];
  __shared__ int      s_adj[SELF ? 1 : DEG];

  if (!SELF)
    for (int k = tid; k < DEG; k += NTHR) s_adj[k] = adj[(size_t)node * DEG + k];
  for (int d = tid; d < COUT; d += NTHR) s_sum[d] = 0.f;
  __syncthreads();

  // ---- weights: raw f32 -> Wt (transposed f16), per-column sum of squares ----
  {
    const float* vp = v + (size_t)node * CIN * COUT;
    const int dcol = tid % COUT;           // constant per thread
    float lsum = 0.f;
    for (int idx = tid; idx < CIN * COUT; idx += NTHR) {
      const int c = idx / COUT;            // idx % COUT == dcol
      const float f = vp[idx];
      Wt[dcol][c] = (_Float16)f;
      lsum += f * f;
    }
    atomicAdd(&s_sum[dcol], lsum);         // ds_add_f32, 2-way contention
  }

  // ---- H: mean over gathered f16 blocks (u32 packed-pair loads), or self ----
  {
    const unsigned int* zu = (const unsigned int*)zin;
    for (int p = tid; p < HNP; p += NTHR) {
      float a0, a1;
      if (SELF) {
        HalfPair hp; hp.u = zu[(size_t)node * HNP + p];
        a0 = (float)hp.h[0]; a1 = (float)hp.h[1];
      } else {
        a0 = 0.f; a1 = 0.f;
        #pragma unroll 4
        for (int k = 0; k < DEG; ++k) {
          HalfPair hp; hp.u = zu[(size_t)s_adj[k] * HNP + p];
          a0 += (float)hp.h[0]; a1 += (float)hp.h[1];
        }
        a0 *= (1.0f / (float)DEG); a1 *= (1.0f / (float)DEG);
      }
      const int e = p * 2;                 // CIN even => pair stays in one row
      const int row = e / CIN, c = e % CIN;
      Hh[row][c]     = (_Float16)a0;
      Hh[row][c + 1] = (_Float16)a1;
    }
  }
  __syncthreads();

  for (int d = tid; d < COUT; d += NTHR)
    s_scale[d] = g[(size_t)node * COUT + d] * rsqrtf(s_sum[d] + EPSF);
  __syncthreads();

  // ---- apply weight-norm scale in place ----
  for (int idx = tid; idx < CIN * COUT; idx += NTHR) {
    const int d = idx / CIN, c = idx % CIN;
    Wt[d][c] = (_Float16)((float)Wt[d][c] * s_scale[d]);
  }
  __syncthreads();

  // ---- WMMA compute: wave = N-tile; 2 M-tiles; KCH chunks ----
  const int wid   = tid >> 5;          // N-tile index
  const int lane  = tid & 31;
  const int lhalf = lane >> 4;         // 0: lanes 0-15, 1: lanes 16-31
  const int lmod  = lane & 15;
  const int colg  = wid * 16 + lmod;   // global output column for B/C/D
  const float bias = b[(size_t)node * COUT + colg];

  #pragma unroll
  for (int mtile = 0; mtile < 2; ++mtile) {
    Frag8f acc;
    #pragma unroll
    for (int r = 0; r < 8; ++r) acc.f[r] = bias;   // bias constant per column

    #pragma unroll
    for (int kc = 0; kc < KCH; ++kc) {
      Frag16 A, B;
      // A (16-bit 16x32): lane(0-15)=row, holds K{kb..kb+7, kb+16..kb+23}
      const int m  = mtile * 16 + lmod;
      const int ka = kc * 32 + lhalf * 8;
      #pragma unroll
      for (int i = 0; i < 8; ++i) A.h[i]     = Hh[m][ka + i];
      #pragma unroll
      for (int i = 0; i < 8; ++i) A.h[8 + i] = Hh[m][ka + 16 + i];
      // B (16-bit 32x16): lane = column, holds K{kb..kb+7, kb+16..kb+23}
      const int kb = kc * 32 + lhalf * 8;
      #pragma unroll
      for (int i = 0; i < 8; ++i) B.h[i]     = Wt[colg][kb + i];
      #pragma unroll
      for (int i = 0; i < 8; ++i) B.h[8 + i] = Wt[colg][kb + 16 + i];

      acc.v = __builtin_amdgcn_wmma_f32_16x16x32_f16(
          false, A.v, false, B.v, (short)0, acc.v, false, false);
    }

    // C/D f32 16x16: lane = column, VGPR r = row (mtile*16 + lhalf*8 + r)
    TOUT* zo = zout + (size_t)node * 32 * COUT;
    #pragma unroll
    for (int r = 0; r < 8; ++r) {
      const int mrow = mtile * 16 + lhalf * 8 + r;
      const size_t o = (size_t)mrow * COUT + colg;
      float val = acc.f[r];
      if (ADDIN) val += (float)Hh[mrow][colg];   // fused residual from LDS
      zo[o] = (TOUT)val;
    }
  }
}

// ---------------------------------------------------------------------------
// FC head: out[n,k] = g_k * rsqrt(||fc_v[k]||^2 + eps) * <fc_v[k], flat[n]> + b_k
// flat[n, d*128 + j] = z3[j][n][d]. One block per (k, n); fused sumsq + dot.
// z3 is 2 MB (L2-resident); fc_v rows re-read 32x but also L2-resident.
// ---------------------------------------------------------------------------
__global__ __launch_bounds__(256) void fc_kernel(
    const float* __restrict__ z3, const float* __restrict__ fc_v,
    const float* __restrict__ fc_g, const float* __restrict__ fc_b,
    float* __restrict__ out)
{
  const int k = blockIdx.x, n = blockIdx.y, tid = threadIdx.x;
  __shared__ float s_dot[256];
  __shared__ float s_ss[256];
  float dot = 0.f, ss = 0.f;
  const float* wv = fc_v + (size_t)k * 16384;
  for (int i = tid; i < 16384; i += 256) {
    const float w = wv[i];
    ss += w * w;
    const int d = i >> 7, j = i & 127;
    dot += w * z3[(size_t)j * 4096 + (size_t)n * 128 + d];
  }
  s_dot[tid] = dot; s_ss[tid] = ss;
  __syncthreads();
  for (int s = 128; s > 0; s >>= 1) {
    if (tid < s) { s_dot[tid] += s_dot[tid + s]; s_ss[tid] += s_ss[tid + s]; }
    __syncthreads();
  }
  if (tid == 0)
    out[(size_t)n * 20 + k] = fc_g[k] * rsqrtf(s_ss[0] + EPSF) * s_dot[0] + fc_b[k];
}

// ---------------------------------------------------------------------------
extern "C" void kernel_launch(void* const* d_in, const int* in_sizes, int n_in,
                              void* d_out, int out_size, void* d_ws, size_t ws_size,
                              hipStream_t stream) {
  (void)in_sizes; (void)n_in; (void)out_size; (void)ws_size;

  const float* x    = (const float*)d_in[0];
  const int*   adj0 = (const int*)  d_in[1];
  const float* v0   = (const float*)d_in[2];
  const float* g0   = (const float*)d_in[3];
  const float* b0   = (const float*)d_in[4];
  const int*   adj1 = (const int*)  d_in[5];
  const float* v1   = (const float*)d_in[6];
  const float* g1   = (const float*)d_in[7];
  const float* b1   = (const float*)d_in[8];
  const int*   adj2 = (const int*)  d_in[9];   // self adjacency (unused)
  const float* v2   = (const float*)d_in[10];
  const float* g2   = (const float*)d_in[11];
  const float* b2   = (const float*)d_in[12];
  const int*   adj3 = (const int*)  d_in[13];
  const float* v3   = (const float*)d_in[14];
  const float* g3   = (const float*)d_in[15];
  const float* b3   = (const float*)d_in[16];
  const float* fc_v = (const float*)d_in[17];
  const float* fc_g = (const float*)d_in[18];
  const float* fc_b = (const float*)d_in[19];
  float* out = (float*)d_out;

  char* ws = (char*)d_ws;
  const size_t Z0B = (size_t)65536 * 1024 * sizeof(_Float16);  // 134 MB: z0 f16 [65536][32][32]
  const size_t Z1B = (size_t)16384 * 2048 * sizeof(_Float16);  //  67 MB: z1/t f16 [16384][32][64]
  _Float16* z0 = (_Float16*)ws;
  _Float16* z1 = (_Float16*)(ws + Z0B);
  float*    z3 = (float*)(ws + Z0B + Z1B);                     //   2 MB: z3 f32 [128][32][128]

  // L0: outer product (Cin=1), f16 out
  fgl0_kernel<<<65536, 128, 0, stream>>>(x, adj0, v0, g0, b0, z0);
  // L1: gather-4 + 32x32 @ 32x64 per node, f16 out
  fgl_wmma_kernel<32, 64, 4, false, false, _Float16>
      <<<16384, 128, 0, stream>>>(z0, adj1, v1, g1, b1, z1);
  // L2: self-adj 32x64 @ 64x64, fused residual (from LDS), in place f16
  fgl_wmma_kernel<64, 64, 1, true, true, _Float16>
      <<<16384, 128, 0, stream>>>(z1, adj2, v2, g2, b2, z1);
  // L3: gather-128 + 32x64 @ 64x128 per node (source L2-resident), f32 out
  fgl_wmma_kernel<64, 128, 128, false, false, float>
      <<<128, 256, 0, stream>>>(z1, adj3, v3, g3, b3, z3);
  // FC head
  fc_kernel<<<dim3(20, 32), 256, 0, stream>>>(z3, fc_v, fc_g, fc_b, out);
}
